// Mamba3SISOBlock_38869454029159
// MI455X (gfx1250) — matrix operand order repaired and checked by reading, not compiled, not run on hardware
//
#include <hip/hip_runtime.h>
#include <hip/hip_bf16.h>

// ---------------------------------------------------------------------------
// Mamba SSM block for gfx1250 (MI455X).  bf16 WMMA GEMMs with LDS-staged,
// double-buffered A tiles and one-step-pipelined B fragments; wave-parallel
// selective scan; fused conv+SiLU; fused gating.
// ---------------------------------------------------------------------------

#define D_MODEL 1024
#define D_STATE 16
#define D_CONV  4
#define D_INNER 2048
#define BATCH   2
#define SEQLEN  2048
#define MROWS   (BATCH * SEQLEN)      // 4096
#define XZCOLS  (2 * D_INNER)         // 4096

typedef __attribute__((ext_vector_type(16))) __bf16 v16bf;
typedef __attribute__((ext_vector_type(8)))  __bf16 v8bf;
typedef __attribute__((ext_vector_type(8)))  float  v8f;
typedef __attribute__((ext_vector_type(4)))  float  v4f;

__device__ __forceinline__ v8f wmma_bf16(v16bf a, v16bf b, v8f c) {
    // D = A(16x32 bf16) * B(32x16 bf16) + C(16x16 f32)
    return __builtin_amdgcn_wmma_f32_16x16x32_bf16(
        /*neg_a=*/false, a, /*neg_b=*/false, b,
        /*c_mod=*/(short)0, c, /*reuse_a=*/false, /*reuse_b=*/false);
}

// ---------------------------------------------------------------------------
// Weight prep: fp32 (K x N, row major) -> bf16 transposed (N x K).
// ---------------------------------------------------------------------------
__global__ void cvt_transpose_kernel(const float* __restrict__ src,
                                     __bf16* __restrict__ dst,
                                     int N, int kbits) {
    unsigned id = blockIdx.x * 256u + threadIdx.x;   // over N*K
    unsigned K = 1u << kbits;
    unsigned k = id & (K - 1u);
    unsigned n = id >> kbits;
    dst[id] = (__bf16)src[(size_t)k * N + n];
}

// ---------------------------------------------------------------------------
// bf16 WMMA GEMM:  C(f32, M x ldc) = A(f32, M x lda, cvt->bf16) * BT(N x K).
// Block = 128x128 tile, 8 waves, wave tile 64x32 (4x2 WMMA accumulators).
// A tile (128x32) staged fp32->bf16 into LDS, double buffered; B fragments
// pipelined one k-step ahead so no full load-drain at the barrier:
//   [cvt+ds_store(buf) ; issue A(k+1) ; issue B(k+1) ; barrier ;
//    ds_load frags + WMMA with B(k)]
// LDS row stride = 40 bf16 (80 B): 16B-aligned b128 access, 20*r mod 64
// distinct for r=0..15 -> conflict-free fragment reads.
// ---------------------------------------------------------------------------
#define ASTRIDE 40

__global__ __launch_bounds__(256)
void wmma_gemm_kernel(const float* __restrict__ A,
                      const __bf16* __restrict__ BT,
                      float* __restrict__ C,
                      int lda, int ldc, int K) {
    __shared__ __align__(16) __bf16 sA[2][128 * ASTRIDE];   // 20 KB

    const int tid  = threadIdx.x;
    const int lane = tid & 31;
    const int wv   = tid >> 5;          // 0..7
    const int half = lane >> 4;         // 0/1
    const int r    = lane & 15;
    const size_t mblk = (size_t)blockIdx.y * 128;
    const size_t n0   = (size_t)blockIdx.x * 128 + (size_t)(wv & 3) * 32;
    const int    mw   = (wv >> 2) * 64; // wave row offset inside tile

    // Staging coords: thread -> (row srow, 16-float chunk skk). Coalesced.
    const int srow = tid >> 1;
    const int skk  = (tid & 1) * 16;
    const float* sgp = A + (mblk + srow) * (size_t)lda + skk;
    __bf16* sdst = &sA[0][0] + srow * ASTRIDE + skk;        // buf 0 base

    const __bf16* brow[2];
#pragma unroll
    for (int j = 0; j < 2; ++j)
        brow[j] = BT + (n0 + (size_t)j * 16 + r) * (size_t)K + half * 16;

    // Fragment LDS element offsets: frag mt reads row mw+mt*16+r.
    int fbase[4];
#pragma unroll
    for (int mt = 0; mt < 4; ++mt)
        fbase[mt] = (mw + mt * 16 + r) * ASTRIDE + half * 16;

    v8f acc[4][2];
#pragma unroll
    for (int i = 0; i < 4; ++i)
#pragma unroll
        for (int j = 0; j < 2; ++j)
#pragma unroll
            for (int e = 0; e < 8; ++e) acc[i][j][e] = 0.f;

    // Preload k0 = 0: raw A chunk and B fragments.
    v4f r0 = *(const v4f*)(sgp);
    v4f r1 = *(const v4f*)(sgp + 4);
    v4f r2 = *(const v4f*)(sgp + 8);
    v4f r3 = *(const v4f*)(sgp + 12);
    v16bf bcur0 = *(const v16bf*)(brow[0]);
    v16bf bcur1 = *(const v16bf*)(brow[1]);

    int buf = 0;
#pragma unroll 2
    for (int k0 = 0; k0 < K; k0 += 32) {
        // ---- convert current raw A chunk, store to LDS[buf] ----
        v8bf h0, h1;
#pragma unroll
        for (int i = 0; i < 4; ++i) {
            h0[i]     = (__bf16)r0[i];
            h0[4 + i] = (__bf16)r1[i];
            h1[i]     = (__bf16)r2[i];
            h1[4 + i] = (__bf16)r3[i];
        }
        __bf16* d = sdst + buf * (128 * ASTRIDE);
        *(v8bf*)(d)     = h0;
        *(v8bf*)(d + 8) = h1;

        // ---- issue NEXT k-step loads (A raw first, then B fragments) ----
        v16bf bnext0 = bcur0, bnext1 = bcur1;
        if (k0 + 32 < K) {
            const float* np = sgp + k0 + 32;
            r0 = *(const v4f*)(np);
            r1 = *(const v4f*)(np + 4);
            r2 = *(const v4f*)(np + 8);
            r3 = *(const v4f*)(np + 12);
            bnext0 = *(const v16bf*)(brow[0] + k0 + 32);
            bnext1 = *(const v16bf*)(brow[1] + k0 + 32);
            __builtin_prefetch(brow[0] + k0 + 64, 0, 3);
            __builtin_prefetch(brow[1] + k0 + 64, 0, 3);
        }

        __syncthreads();   // LDS[buf] fully staged by all waves

        // ---- fragment reads from LDS + 8x WMMA with B(k0) ----
        const __bf16* sb = &sA[0][0] + buf * (128 * ASTRIDE);
#pragma unroll
        for (int mt = 0; mt < 4; ++mt) {
            v8bf lo = *(const v8bf*)(sb + fbase[mt]);
            v8bf hi = *(const v8bf*)(sb + fbase[mt] + 8);
            v16bf af = __builtin_shufflevector(lo, hi,
                0, 1, 2, 3, 4, 5, 6, 7, 8, 9, 10, 11, 12, 13, 14, 15);
            acc[mt][0] = wmma_bf16(af, bcur0, acc[mt][0]);
            acc[mt][1] = wmma_bf16(af, bcur1, acc[mt][1]);
        }
        bcur0 = bnext0;
        bcur1 = bnext1;
        buf ^= 1;
    }

    // C layout: lane col = n0 + j*16 + (lane&15); VGPR e -> row e + 8*half.
#pragma unroll
    for (int mt = 0; mt < 4; ++mt) {
        size_t rbase = mblk + mw + mt * 16 + half * 8;
#pragma unroll
        for (int j = 0; j < 2; ++j) {
            size_t col = n0 + (size_t)j * 16 + r;
#pragma unroll
            for (int e = 0; e < 8; ++e)
                C[(rbase + e) * (size_t)ldc + col] = acc[mt][j][e];
        }
    }
}

// ---------------------------------------------------------------------------
// Depthwise causal conv(4) + bias + SiLU on x_branch (first half of xz).
// ---------------------------------------------------------------------------
__global__ __launch_bounds__(256)
void conv_silu_kernel(const float* __restrict__ xz,
                      const float* __restrict__ cw,
                      const float* __restrict__ cb,
                      float* __restrict__ xc) {
    int id = blockIdx.x * 256 + threadIdx.x;   // over MROWS*D_INNER
    int c  = id & (D_INNER - 1);
    int m  = id >> 11;
    int l  = m & (SEQLEN - 1);
    float acc = cb[c];
    const float* w = cw + c * D_CONV;
#pragma unroll
    for (int j = 0; j < D_CONV; ++j) {
        int li = l + j - (D_CONV - 1);
        if (li >= 0)
            acc = fmaf(w[j], xz[(size_t)(m + j - (D_CONV - 1)) * XZCOLS + c], acc);
    }
    float s = acc / (1.f + __expf(-acc));      // SiLU
    xc[(size_t)m * D_INNER + c] = s;
}

// ---------------------------------------------------------------------------
// params = x_conv @ W_param (2048 x 33).  4 rows staged in LDS per block;
// wave-level dot products, lane-strided K for conflict-free LDS reads.
// ---------------------------------------------------------------------------
__global__ __launch_bounds__(256)
void params_kernel(const float* __restrict__ xc,
                   const float* __restrict__ Wp,
                   float* __restrict__ P) {
    __shared__ float lds[4 * D_INNER];
    const int tid = threadIdx.x;
    const int m0  = blockIdx.x * 4;
    for (int i = tid; i < 4 * D_INNER; i += 256)
        lds[i] = xc[(size_t)m0 * D_INNER + i];
    __syncthreads();
    const int w = tid >> 5, lane = tid & 31;
    for (int task = w; task < 4 * 33; task += 8) {
        int rr = task / 33, j = task % 33;
        const float* row = lds + rr * D_INNER;
        float p = 0.f;
#pragma unroll 4
        for (int i = 0; i < D_INNER / 32; ++i) {
            int k = (i << 5) + lane;
            p = fmaf(row[k], Wp[k * 33 + j], p);
        }
        p += __shfl_xor(p, 16);
        p += __shfl_xor(p, 8);
        p += __shfl_xor(p, 4);
        p += __shfl_xor(p, 2);
        p += __shfl_xor(p, 1);
        if (lane == 0) P[(size_t)(m0 + rr) * 33 + j] = p;
    }
}

// ---------------------------------------------------------------------------
// Selective scan, fused with D-skip and SiLU(z) gating.
// 16 lanes per channel, one state per lane; width-16 shfl-xor reduction
// performs the einsum over states each timestep.  2048 waves in flight.
// ---------------------------------------------------------------------------
__global__ __launch_bounds__(256)
void scan_kernel(const float* __restrict__ P,     // 4096 x 33
                 const float* __restrict__ xc,    // 4096 x 2048
                 const float* __restrict__ xz,    // 4096 x 4096 (z in cols 2048+)
                 const float* __restrict__ wdt,
                 const float* __restrict__ bdt,
                 const float* __restrict__ Alog,
                 const float* __restrict__ Dp,
                 float* __restrict__ yg) {
    const int tid = threadIdx.x;
    const int n   = tid & 15;
    const int c   = (blockIdx.x & 127) * 16 + (tid >> 4);
    const int b   = blockIdx.x >> 7;
    const float A  = -__expf(Alog[c * D_STATE + n]);
    const float wd = wdt[c], bd = bdt[c], Dc = Dp[c];
    float h = 0.f;
    const size_t base = (size_t)b * SEQLEN;
    for (int t = 0; t < SEQLEN; ++t) {
        size_t m = base + t;
        const float* pr = P + m * 33;
        float s  = pr[0];
        float Bt = pr[1 + n];
        float Ct = pr[17 + n];
        float xv = xc[m * D_INNER + c];
        float v  = fmaf(s, wd, bd);
        float dt = (v > 30.f) ? v : __logf(1.f + __expf(v));   // softplus
        h = fmaf(__expf(A * dt), h, dt * Bt * xv);
        float p = h * Ct;
        p += __shfl_xor(p, 8, 16);
        p += __shfl_xor(p, 4, 16);
        p += __shfl_xor(p, 2, 16);
        p += __shfl_xor(p, 1, 16);
        if (n == 0) {
            float z = xz[m * XZCOLS + D_INNER + c];
            float g = z / (1.f + __expf(-z));                  // SiLU(z)
            yg[m * D_INNER + c] = (p + Dc * xv) * g;
        }
    }
}

// ---------------------------------------------------------------------------
// Workspace layout (bytes)
// ---------------------------------------------------------------------------
#define OFF_XZ     ((size_t)0)                               // 4096*4096 f32
#define OFF_XCONV  (OFF_XZ    + (size_t)MROWS * XZCOLS * 4)  // 4096*2048 f32
#define OFF_YGATE  (OFF_XCONV + (size_t)MROWS * D_INNER * 4)
#define OFF_PARAMS (OFF_YGATE + (size_t)MROWS * D_INNER * 4) // 4096*33 f32
#define OFF_WTIN   (OFF_PARAMS + (size_t)MROWS * 33 * 4)     // 4096*1024 bf16
#define OFF_WTOUT  (OFF_WTIN  + (size_t)XZCOLS * D_MODEL * 2)// 1024*2048 bf16

extern "C" void kernel_launch(void* const* d_in, const int* in_sizes, int n_in,
                              void* d_out, int out_size, void* d_ws, size_t ws_size,
                              hipStream_t stream) {
    const float* x      = (const float*)d_in[0];
    const float* W_in   = (const float*)d_in[1];
    const float* conv_w = (const float*)d_in[2];
    const float* conv_b = (const float*)d_in[3];
    const float* W_prm  = (const float*)d_in[4];
    const float* w_dt   = (const float*)d_in[5];
    const float* b_dt   = (const float*)d_in[6];
    const float* A_log  = (const float*)d_in[7];
    const float* D_prm  = (const float*)d_in[8];
    const float* W_out  = (const float*)d_in[9];

    char* ws = (char*)d_ws;
    float*  xz     = (float*)(ws + OFF_XZ);
    float*  xconv  = (float*)(ws + OFF_XCONV);
    float*  ygate  = (float*)(ws + OFF_YGATE);
    float*  params = (float*)(ws + OFF_PARAMS);
    __bf16* WT_in  = (__bf16*)(ws + OFF_WTIN);
    __bf16* WT_out = (__bf16*)(ws + OFF_WTOUT);

    // 0) weight transpose+convert to bf16
    cvt_transpose_kernel<<<(XZCOLS * D_MODEL) / 256, 256, 0, stream>>>(
        W_in, WT_in, /*N=*/XZCOLS, /*kbits=*/10);               // K=1024
    cvt_transpose_kernel<<<(D_MODEL * D_INNER) / 256, 256, 0, stream>>>(
        W_out, WT_out, /*N=*/D_MODEL, /*kbits=*/11);            // K=2048

    // 1) xz = x @ W_in   (4096x1024 * 1024x4096)
    {
        dim3 grid(XZCOLS / 128, MROWS / 128);
        wmma_gemm_kernel<<<grid, 256, 0, stream>>>(
            x, WT_in, xz, /*lda=*/D_MODEL, /*ldc=*/XZCOLS, /*K=*/D_MODEL);
    }

    // 2) depthwise conv + SiLU
    conv_silu_kernel<<<(MROWS * D_INNER) / 256, 256, 0, stream>>>(
        xz, conv_w, conv_b, xconv);

    // 3) params = x_conv @ W_param  (N=33, wave-dot path)
    params_kernel<<<MROWS / 4, 256, 0, stream>>>(xconv, W_prm, params);

    // 4) selective scan + D-skip + gating
    scan_kernel<<<BATCH * (D_INNER / 16), 256, 0, stream>>>(
        params, xconv, xz, w_dt, b_dt, A_log, D_prm, ygate);

    // 5) out = y_gated @ W_out  (4096x2048 * 2048x1024)
    {
        dim3 grid(D_MODEL / 128, MROWS / 128);
        wmma_gemm_kernel<<<grid, 256, 0, stream>>>(
            ygate, WT_out, (float*)d_out, /*lda=*/D_INNER, /*ldc=*/D_MODEL,
            /*K=*/D_INNER);
    }
}
